// SparseTopKAutoEncoder_38328288150205
// MI455X (gfx1250) — compile-verified, matrix operand order checked by reading, not run on hardware
//
#include <hip/hip_runtime.h>
#include <hip/hip_bf16.h>

#define D_IN   1024
#define D_HID  16384
#define TOPK   32
#define BATCH  8192

// encoder tiling
#define KT   32
#define BM   64
#define BN   256
#define LDA  40   // padded bf16 row stride (80 B, multiple of 16 B for b128 loads)
#define NKT  (D_IN / KT)   // 32 k-steps

typedef __attribute__((ext_vector_type(16))) __bf16 v16bf;
typedef __attribute__((ext_vector_type(8)))  __bf16 v8bf;
typedef __attribute__((ext_vector_type(8)))  float  v8f;

// Build a 16-element bf16 fragment from two aligned 16-byte LDS loads.
// ISA 16-bit 16x32 layout: lane group g holds K=g*8..g*8+7 then K=16+g*8..+7.
static __device__ inline v16bf mk_frag(const __bf16* p) {
    v8bf a = *(const v8bf*)p;
    v8bf b = *(const v8bf*)(p + 16);
    v16bf r;
#pragma unroll
    for (int i = 0; i < 8; ++i) { r[i] = a[i]; r[i + 8] = b[i]; }
    return r;
}

static __device__ inline unsigned pack_bf16(float f0, float f1) {
    union { __bf16 b[2]; unsigned u; } p;
    p.b[0] = (__bf16)f0;
    p.b[1] = (__bf16)f1;
    return p.u;
}

__global__ __launch_bounds__(256)
void enc_kernel(const float* __restrict__ x, const float* __restrict__ We,
                const float* __restrict__ be, float* __restrict__ h)
{
    // [buf][hi=0/lo=1][...]
    __shared__ __align__(16) __bf16 sA[2][2][BM * LDA];   // A staged [M][K]
    __shared__ __align__(16) __bf16 sB[2][2][BN * LDA];   // B staged transposed [N][K]

    const int tid  = threadIdx.x;
    const int lane = tid & 31;
    const int wave = tid >> 5;
    const int wm0  = (wave & 1) * 32;    // wave M offset in block tile (2 waves on M)
    const int wn0  = (wave >> 1) * 64;   // wave N offset in block tile (4 waves on N)
    const int m0   = blockIdx.y * BM;
    const int n0   = blockIdx.x * BN;
    const int g    = lane >> 4;          // lane group
    const int ml   = lane & 15;          // M (A) / N (B) within 16-subtile

    float2 areg[4];     // register staging: A tile 64x32 -> 4 float2 / thread
    float  breg[32];    // register staging: B tile 32x256 -> 32 floats / thread

    auto load_tile = [&](int k0) {
#pragma unroll
        for (int i = 0; i < 4; ++i) {
            int p = tid + 256 * i;           // 1024 K-pair slots (16 pairs x 64 rows)
            int r = p >> 4, kp = p & 15;     // coalesced along K within a row
            areg[i] = *(const float2*)(x + (size_t)(m0 + r) * D_IN + (k0 + 2 * kp));
        }
#pragma unroll
        for (int i = 0; i < 16; ++i) {
            int p = tid + 256 * i;           // 4096 (kpair, n) slots
            int kp = p >> 8, nn = p & 255;   // coalesced along N
            const float* base = We + (size_t)(k0 + 2 * kp) * D_HID + (n0 + nn);
            breg[2 * i]     = base[0];
            breg[2 * i + 1] = base[D_HID];
        }
    };

    auto store_tile = [&](int buf) {
#pragma unroll
        for (int i = 0; i < 4; ++i) {
            int p = tid + 256 * i;
            int r = p >> 4, kp = p & 15;
            float f0 = areg[i].x, f1 = areg[i].y;
            __bf16 h0 = (__bf16)f0, h1 = (__bf16)f1;
            float  l0 = f0 - (float)h0, l1 = f1 - (float)h1;
            *(unsigned*)&sA[buf][0][r * LDA + 2 * kp] = pack_bf16(f0, f1);
            *(unsigned*)&sA[buf][1][r * LDA + 2 * kp] = pack_bf16(l0, l1);
        }
#pragma unroll
        for (int i = 0; i < 16; ++i) {
            int p = tid + 256 * i;
            int kp = p >> 8, nn = p & 255;
            float f0 = breg[2 * i], f1 = breg[2 * i + 1];
            __bf16 h0 = (__bf16)f0, h1 = (__bf16)f1;
            float  l0 = f0 - (float)h0, l1 = f1 - (float)h1;
            *(unsigned*)&sB[buf][0][nn * LDA + 2 * kp] = pack_bf16(f0, f1);
            *(unsigned*)&sB[buf][1][nn * LDA + 2 * kp] = pack_bf16(l0, l1);
        }
    };

    v8f acc[2][4] = {};

    // prologue: stage tile 0
    load_tile(0);
    store_tile(0);
    __syncthreads();

    int cur = 0;
    for (int t = 0; t < NKT; ++t) {
        // prefetch next tile into registers (hidden behind the WMMAs below)
        if (t + 1 < NKT) load_tile((t + 1) * KT);

        // A fragments for this wave: 2 M-subtiles, hi & lo
        v16bf ah[2], al[2];
#pragma unroll
        for (int mt = 0; mt < 2; ++mt) {
            const int ro = (wm0 + mt * 16 + ml) * LDA + g * 8;
            ah[mt] = mk_frag(&sA[cur][0][ro]);
            al[mt] = mk_frag(&sA[cur][1][ro]);
        }

#pragma unroll
        for (int nt = 0; nt < 4; ++nt) {
            const int ro = (wn0 + nt * 16 + ml) * LDA + g * 8;
            v16bf bh = mk_frag(&sB[cur][0][ro]);
            v16bf bl = mk_frag(&sB[cur][1][ro]);
#pragma unroll
            for (int mt = 0; mt < 2; ++mt) {
                // split-float f32 emulation: hi*lo + lo*hi + hi*hi
                acc[mt][nt] = __builtin_amdgcn_wmma_f32_16x16x32_bf16(
                    false, ah[mt], false, bl, (short)0, acc[mt][nt], false, false);
                acc[mt][nt] = __builtin_amdgcn_wmma_f32_16x16x32_bf16(
                    false, al[mt], false, bh, (short)0, acc[mt][nt], false, false);
                acc[mt][nt] = __builtin_amdgcn_wmma_f32_16x16x32_bf16(
                    false, ah[mt], false, bh, (short)0, acc[mt][nt], false, false);
            }
        }

        if (t + 1 < NKT) {
            store_tile(cur ^ 1);    // convert + fill the other buffer
            __syncthreads();
            cur ^= 1;
        }
    }

    // epilogue: bias + ReLU, write dense h
    // C layout: element r -> M = r + 8*g, N = lane&15
#pragma unroll
    for (int nt = 0; nt < 4; ++nt) {
        int gn = n0 + wn0 + nt * 16 + ml;
        float bias = be[gn];
#pragma unroll
        for (int mt = 0; mt < 2; ++mt) {
#pragma unroll
            for (int r = 0; r < 8; ++r) {
                int gm = m0 + wm0 + mt * 16 + 8 * g + r;
                float vv = acc[mt][nt][r] + bias;
                h[(size_t)gm * D_HID + gn] = vv > 0.0f ? vv : 0.0f;
            }
        }
    }
}

__global__ __launch_bounds__(256)
void topk_kernel(float* __restrict__ h, float* __restrict__ scal,
                 int* __restrict__ tki, float* __restrict__ tkv)
{
    __shared__ float rv[256];
    __shared__ int   rc[256];
    __shared__ float tv[TOPK];
    __shared__ int   tc[TOPK];

    const int tid = threadIdx.x;
    const size_t row = blockIdx.x;
    float* hrow = h + row * (size_t)D_HID;

    float v[64];
#pragma unroll
    for (int i = 0; i < 64; ++i) v[i] = hrow[i * 256 + tid];  // coalesced

    for (int t = 0; t < TOPK; ++t) {
        float lm = -1.0f; int li = 0;        // relu output >= 0 > sentinel
#pragma unroll
        for (int i = 0; i < 64; ++i) {
            if (v[i] > lm) { lm = v[i]; li = i; }
        }
        rv[tid] = lm;
        rc[tid] = li * 256 + tid;            // global column
        __syncthreads();
        for (int s = 128; s > 0; s >>= 1) {
            if (tid < s) {
                float ov = rv[tid + s]; int oc = rc[tid + s];
                if (ov > rv[tid] || (ov == rv[tid] && oc < rc[tid])) {
                    rv[tid] = ov; rc[tid] = oc;
                }
            }
            __syncthreads();
        }
        int wc = rc[0];
        if ((wc & 255) == tid) {
            int widx = wc >> 8;
#pragma unroll
            for (int i = 0; i < 64; ++i)     // static indices -> stays in VGPRs
                if (i == widx) v[i] = -1.0f;
        }
        if (tid == 0) { tv[t] = rv[0]; tc[t] = wc; }
        __syncthreads();
    }

    // rewrite row: zeros everywhere, then scatter the kept top-k values
#pragma unroll
    for (int i = 0; i < 64; ++i) hrow[i * 256 + tid] = 0.0f;
    __syncthreads();
    if (tid < TOPK) {
        hrow[tc[tid]] = tv[tid];
        tki[row * TOPK + tid] = tc[tid];
        tkv[row * TOPK + tid] = tv[tid];
    }
    if (tid == 0) {
        float s1 = 0.0f, s0 = 0.0f;
        for (int t = 0; t < TOPK; ++t) { s1 += tv[t]; if (tv[t] > 0.0f) s0 += 1.0f; }
        const float inv = 1.0f / ((float)BATCH * (float)D_HID);
        atomicAdd(&scal[2], s1 * inv);  // l1_loss
        atomicAdd(&scal[3], s0 * inv);  // l0_loss
    }
}

__global__ __launch_bounds__(256)
void dec_kernel(const float* __restrict__ x, const float* __restrict__ Wd,
                const float* __restrict__ bd, const int* __restrict__ tki,
                const float* __restrict__ tkv, float* __restrict__ dec,
                float* __restrict__ scal)
{
    __shared__ int   si[TOPK];
    __shared__ float sv[TOPK];
    __shared__ float red[256];

    const int tid = threadIdx.x;
    const size_t row = blockIdx.x;
    if (tid < TOPK) {
        si[tid] = tki[row * TOPK + tid];
        sv[tid] = tkv[row * TOPK + tid];
    }
    __syncthreads();

    const int c = tid * 4;
    float4 a = *(const float4*)(bd + c);
#pragma unroll 8
    for (int t = 0; t < TOPK; ++t) {
        const float4 w = *(const float4*)(Wd + (size_t)si[t] * D_IN + c);
        const float s = sv[t];
        a.x += s * w.x; a.y += s * w.y; a.z += s * w.z; a.w += s * w.w;
    }
    *(float4*)(dec + row * (size_t)D_IN + c) = a;

    const float4 xe = *(const float4*)(x + row * (size_t)D_IN + c);
    float dx = a.x - xe.x, dy = a.y - xe.y, dz = a.z - xe.z, dw = a.w - xe.w;
    red[tid] = dx * dx + dy * dy + dz * dz + dw * dw;
    __syncthreads();
    for (int s = 128; s > 0; s >>= 1) {
        if (tid < s) red[tid] += red[tid + s];
        __syncthreads();
    }
    if (tid == 0) {
        const float inv = 1.0f / ((float)BATCH * (float)D_IN);
        float r = red[0] * inv;
        atomicAdd(&scal[0], r);   // recon_loss (slot 1)
        atomicAdd(&scal[1], r);   // recon_loss (slot 2, duplicated in reference)
    }
}

__global__ void init_kernel(float* scal)
{
    if (threadIdx.x < 4) scal[threadIdx.x] = 0.0f;
}

extern "C" void kernel_launch(void* const* d_in, const int* in_sizes, int n_in,
                              void* d_out, int out_size, void* d_ws, size_t ws_size,
                              hipStream_t stream) {
    const float* x  = (const float*)d_in[0];
    const float* We = (const float*)d_in[1];
    const float* be = (const float*)d_in[2];
    const float* Wd = (const float*)d_in[3];
    const float* bd = (const float*)d_in[4];

    float* out  = (float*)d_out;
    float* dec  = out;                                   // [BATCH, D_IN]
    float* h    = out + (size_t)BATCH * D_IN;            // [BATCH, D_HID] (masked)
    float* scal = h + (size_t)BATCH * D_HID;             // [recon, recon, l1, l0]

    int*   tki = (int*)d_ws;                                        // [BATCH, TOPK]
    float* tkv = (float*)((char*)d_ws + (size_t)BATCH * TOPK * 4);  // [BATCH, TOPK]

    init_kernel<<<1, 32, 0, stream>>>(scal);

    dim3 gridE(D_HID / BN, BATCH / BM);   // 64 x 128 blocks
    enc_kernel<<<gridE, 256, 0, stream>>>(x, We, be, h);

    topk_kernel<<<BATCH, 256, 0, stream>>>(h, scal, tki, tkv);

    dec_kernel<<<BATCH, 256, 0, stream>>>(x, Wd, bd, tki, tkv, dec, scal);
}
